// TensorNetInteraction_90314572300301
// MI455X (gfx1250) — matrix operand order repaired and checked by reading, not compiled, not run on hardware
//
#include <hip/hip_runtime.h>
#include <hip/hip_bf16.h>

typedef __attribute__((ext_vector_type(16))) _Float16 v16h;
typedef __attribute__((ext_vector_type(8)))  float    v8f;

#define UNITS   64
#define NUM_RBF 32
#define H3      192

// packed-B sizes (halfs): layer1 4 tiles x 1 chunk, layer2 8x2, layer3 12x4
#define B1_HALFS  (4  * 512)                 //  2048
#define B2_HALFS  (16 * 512)                 //  8192
#define B3_HALFS  (48 * 512)                 // 24576
#define BALL_HALFS (B1_HALFS + B2_HALFS + B3_HALFS)   // 34816 (69632 B)

// fast silu: x * rcp(1+exp(-x)) -- v_exp_f32 + v_rcp_f32, no IEEE div chain
__device__ __forceinline__ float silu_f(float x) {
  return x * __builtin_amdgcn_rcpf(1.0f + __expf(-x));
}

// ---------------------------------------------------------------- zero scratch
__global__ void zero_f32_kernel(float* __restrict__ p, int n) {
  int i = blockIdx.x * blockDim.x + threadIdx.x;
  if (i < n) p[i] = 0.0f;
}

// ------------------------------------------------- pack W[Ncols x K] -> WMMA B
// B element (k, n) = W[n*K + k].  Per 16-col tile t, 32-K chunk c:
//   linear = ((t*nchunk + c)*32 + lane)*16 + h
//   lane: n = t*16 + lane%16 ; k = c*32 + (h%8) + 8*(lane/16) + 16*(h/8)      (f16)
__global__ void pack_b_kernel(const float* __restrict__ W, _Float16* __restrict__ out,
                              int ntile, int nchunk, int K) {
  int i = blockIdx.x * blockDim.x + threadIdx.x;
  int total = ntile * nchunk * 512;
  if (i >= total) return;
  int h    = i & 15;
  int lane = (i >> 4) & 31;
  int tc   = i >> 9;
  int c = tc % nchunk;
  int t = tc / nchunk;
  int n = t * 16 + (lane & 15);
  int k = c * 32 + (h & 7) + 8 * (lane >> 4) + 16 * (h >> 3);
  out[i] = (_Float16)W[n * K + k];
}

// ------------------------------------------------------- fused edge MLP kernel
// 4 waves/block, each wave owns TWO 16-row A subtiles (32 edges) so every B
// tile feeds two WMMAs (register blocking on B).  All packed B weights are
// staged global->LDS once per block with async b128 copies (ASYNCcnt path),
// overlapped with the A1 pack + edge-metadata loads.
// Layer1: K=32   Layer2: K=64   Layer3: K=128 -> *cutoff -> atomicAdd
#define EW  4
#define SUB 2
#define A2_HALFS (EW * SUB * 2 * 512)   //  8192 halfs (16 KB)
#define A3_HALFS (EW * SUB * 4 * 512)   // 16384 halfs (32 KB)
#define SMEM_HALFS (A2_HALFS + A3_HALFS + BALL_HALFS)   // 59392 (116 KB)

__global__ __launch_bounds__(128) void edge_mlp_kernel(
    const float* __restrict__ edge_attr,
    const float* __restrict__ bond_dist,
    const int*   __restrict__ edge_dst,
    const _Float16* __restrict__ Ball,   // contiguous Bp1|Bp2|Bp3
    const float* __restrict__ bs1,
    const float* __restrict__ bs2,
    const float* __restrict__ bs3,
    float* __restrict__ sums,
    int E)
{
  extern __shared__ __attribute__((aligned(64))) _Float16 smem[];
  _Float16* a2s = smem;                       // [EW][SUB][2][32][16]
  _Float16* a3s = smem + A2_HALFS;            // [EW][SUB][4][32][16]
  _Float16* bsh = smem + A2_HALFS + A3_HALFS; // Bp1|Bp2|Bp3 in LDS

  const int lane  = threadIdx.x & 31;
  const int wave  = threadIdx.x >> 5;
  const int lhalf = lane >> 4;     // 0/1 half-wave
  const int lcol  = lane & 15;
  const int base  = blockIdx.x * (EW * SUB * 16) + wave * (SUB * 16);

  // ---- async-stage all packed B weights into LDS (ASYNCcnt) ---------------
  {
    unsigned ldsb = (unsigned)(uintptr_t)bsh;          // LDS byte offset
    for (int i = threadIdx.x; i < (BALL_HALFS * 2) / 16; i += 128) {
      unsigned voff = (unsigned)i * 16u;               // byte offset
      asm volatile("global_load_async_to_lds_b128 %0, %1, %2"
                   :: "v"(ldsb + voff), "v"(voff), "s"(Ball)
                   : "memory");
    }
  }

  // ---- edge metadata loads: latency hides under the async copy ------------
  float ccut[SUB][8]; int dsti[SUB][8];
  #pragma unroll
  for (int s = 0; s < SUB; ++s) {
    #pragma unroll
    for (int r = 0; r < 8; ++r) {
      int row = base + s * 16 + r + 8 * lhalf;
      int rc = row >= E ? E - 1 : row;
      float bd = bond_dist[rc];
      float cc = 0.5f * (__cosf(0.62831853071795864769f * bd) + 1.0f); // pi/5
      ccut[s][r] = bd < 5.0f ? cc : 0.0f;
      dsti[s][r] = edge_dst[rc];
    }
  }

  // ---- A1: edge_attr rows in 16-bit WMMA A layout -------------------------
  // lane: row m = lane%16 ; half h: k = (h%8) + 8*(lane/16) + 16*(h/8)
  v16h a1[SUB];
  #pragma unroll
  for (int s = 0; s < SUB; ++s) {
    int row = base + s * 16 + lcol;
    if (row >= E) row = E - 1;
    const float* p = edge_attr + (size_t)row * NUM_RBF + 8 * lhalf;
    #pragma unroll
    for (int h = 0; h < 8; ++h) {
      a1[s][h]     = (_Float16)p[h];
      a1[s][h + 8] = (_Float16)p[16 + h];
    }
  }

  // all async B copies complete + visible block-wide
  asm volatile("s_wait_asynccnt 0" ::: "memory");
  __syncthreads();

  const _Float16* B1 = bsh;
  const _Float16* B2 = bsh + B1_HALFS;
  const _Float16* B3 = bsh + B1_HALFS + B2_HALFS;

  // ---- layer 1: [16,32] x [32,64] -----------------------------------------
  #pragma unroll
  for (int t = 0; t < 4; ++t) {
    v16h b = *(const v16h*)(B1 + (size_t)(t * 32 + lane) * 16);
    float bias = bs1[t * 16 + lcol];
    int k  = t * 16 + lcol;          // output column 0..63
    int c  = k >> 5, kk = k & 31;    // chunk + k-within-chunk for next layer A
    int dh   = (kk & 7) + 8 * (kk >> 4);
    int dl16 = 16 * ((kk >> 3) & 1);
    #pragma unroll
    for (int s = 0; s < SUB; ++s) {
      v8f acc = {};
      acc = __builtin_amdgcn_wmma_f32_16x16x32_f16(false, a1[s], false, b, (short)0, acc, false, false);
      #pragma unroll
      for (int r = 0; r < 8; ++r) {
        int m = r + 8 * lhalf;       // output row (D layout)
        a2s[((((wave * SUB + s) * 2 + c) * 32) + (m + dl16)) * 16 + dh] =
            (_Float16)silu_f(acc[r] + bias);
      }
    }
  }
  __syncthreads();

  // ---- layer 2: [16,64] x [64,128] ----------------------------------------
  #pragma unroll
  for (int t = 0; t < 8; ++t) {
    v8f acc0 = {}, acc1 = {};
    #pragma unroll
    for (int c = 0; c < 2; ++c) {
      v16h b  = *(const v16h*)(B2 + (size_t)((t * 2 + c) * 32 + lane) * 16);
      v16h a0 = *(const v16h*)(a2s + ((((wave * SUB + 0) * 2 + c) * 32) + lane) * 16);
      v16h a1_ = *(const v16h*)(a2s + ((((wave * SUB + 1) * 2 + c) * 32) + lane) * 16);
      acc0 = __builtin_amdgcn_wmma_f32_16x16x32_f16(false, a0,  false, b, (short)0, acc0, false, false);
      acc1 = __builtin_amdgcn_wmma_f32_16x16x32_f16(false, a1_, false, b, (short)0, acc1, false, false);
    }
    float bias = bs2[t * 16 + lcol];
    int k  = t * 16 + lcol;          // 0..127
    int c  = k >> 5, kk = k & 31;
    int dh   = (kk & 7) + 8 * (kk >> 4);
    int dl16 = 16 * ((kk >> 3) & 1);
    #pragma unroll
    for (int r = 0; r < 8; ++r) {
      int m = r + 8 * lhalf;
      a3s[((((wave * SUB + 0) * 4 + c) * 32) + (m + dl16)) * 16 + dh] =
          (_Float16)silu_f(acc0[r] + bias);
      a3s[((((wave * SUB + 1) * 4 + c) * 32) + (m + dl16)) * 16 + dh] =
          (_Float16)silu_f(acc1[r] + bias);
    }
  }
  __syncthreads();

  // ---- layer 3 + cutoff + scatter-add -------------------------------------
  #pragma unroll
  for (int t = 0; t < 12; ++t) {
    v8f acc0 = {}, acc1 = {};
    #pragma unroll
    for (int c = 0; c < 4; ++c) {
      v16h b  = *(const v16h*)(B3 + (size_t)((t * 4 + c) * 32 + lane) * 16);
      v16h a0 = *(const v16h*)(a3s + ((((wave * SUB + 0) * 4 + c) * 32) + lane) * 16);
      v16h a1_ = *(const v16h*)(a3s + ((((wave * SUB + 1) * 4 + c) * 32) + lane) * 16);
      acc0 = __builtin_amdgcn_wmma_f32_16x16x32_f16(false, a0,  false, b, (short)0, acc0, false, false);
      acc1 = __builtin_amdgcn_wmma_f32_16x16x32_f16(false, a1_, false, b, (short)0, acc1, false, false);
    }
    float bias = bs3[t * 16 + lcol];
    int col = t * 16 + lcol;         // 0..191  ( = unit*3 + component )
    #pragma unroll
    for (int r = 0; r < 8; ++r) {
      int row0 = base + r + 8 * lhalf;
      if (row0 < E) {
        float y = silu_f(acc0[r] + bias) * ccut[0][r];
        atomicAdd(&sums[(size_t)dsti[0][r] * H3 + col], y);
      }
      int row1 = base + 16 + r + 8 * lhalf;
      if (row1 < E) {
        float y = silu_f(acc1[r] + bias) * ccut[1][r];
        atomicAdd(&sums[(size_t)dsti[1][r] * H3 + col], y);
      }
    }
  }
}

// --------------------------------------------------------- fused node kernel
// comps per (n,u): [trace, a01,a02,a12, s00,s01,s02,s11,s12,s22]
// msg[n] = sI*I' + sA*A' + sS*S'  (sums buffer holds the scatter-summed scalars)
__global__ __launch_bounds__(256) void node_kernel(
    const float* __restrict__ X,
    const float* __restrict__ sums,
    const float* __restrict__ Wt0, const float* __restrict__ Wt1,
    const float* __restrict__ Wt2, const float* __restrict__ Wt3,
    const float* __restrict__ Wt4, const float* __restrict__ Wt5,
    float* __restrict__ out, int N)
{
  __shared__ float comps[4][UNITS][10];
  __shared__ float xns[4][UNITS][9];
  const int u  = threadIdx.x & 63;
  const int nl = threadIdx.x >> 6;
  int n = blockIdx.x * 4 + nl;
  bool valid = n < N;
  int nn = valid ? n : N - 1;

  // phase 1: normalize + decompose into structural components
  float x[9];
  {
    const float* xp = X + ((size_t)nn * UNITS + u) * 9;
    float nrm = 0.0f;
    #pragma unroll
    for (int i = 0; i < 9; ++i) { x[i] = xp[i]; nrm += x[i] * x[i]; }
    float inv = 1.0f / (nrm + 1.0f);
    #pragma unroll
    for (int i = 0; i < 9; ++i) { x[i] *= inv; xns[nl][u][i] = x[i]; }
    float tr = (x[0] + x[4] + x[8]) * (1.0f / 3.0f);
    float* cp = comps[nl][u];
    cp[0] = tr;
    cp[1] = 0.5f * (x[1] - x[3]);
    cp[2] = 0.5f * (x[2] - x[6]);
    cp[3] = 0.5f * (x[5] - x[7]);
    cp[4] = x[0] - tr;
    cp[5] = 0.5f * (x[1] + x[3]);
    cp[6] = 0.5f * (x[2] + x[6]);
    cp[7] = x[4] - tr;
    cp[8] = 0.5f * (x[5] + x[7]);
    cp[9] = x[8] - tr;
  }
  __syncthreads();

  // phase 2: channel mix (tlin with Wt0..2 over structural dofs)
  float tp = 0.f, ap0=0.f, ap1=0.f, ap2=0.f;
  float sp0=0.f, sp1=0.f, sp2=0.f, sp3=0.f, sp4=0.f, sp5=0.f;
  #pragma unroll 4
  for (int f = 0; f < UNITS; ++f) {
    float w0 = Wt0[u * UNITS + f];
    float w1 = Wt1[u * UNITS + f];
    float w2 = Wt2[u * UNITS + f];
    const float* cf = comps[nl][f];
    tp  += w0 * cf[0];
    ap0 += w1 * cf[1]; ap1 += w1 * cf[2]; ap2 += w1 * cf[3];
    sp0 += w2 * cf[4]; sp1 += w2 * cf[5]; sp2 += w2 * cf[6];
    sp3 += w2 * cf[7]; sp4 += w2 * cf[8]; sp5 += w2 * cf[9];
  }

  const float* sptr = sums + ((size_t)nn * UNITS + u) * 3;
  float sI = sptr[0], sA = sptr[1], sS = sptr[2];

  float Y[9], M[9];
  Y[0]= tp+sp0;  Y[1]= ap0+sp1; Y[2]= ap1+sp2;
  Y[3]=-ap0+sp1; Y[4]= tp+sp3;  Y[5]= ap2+sp4;
  Y[6]=-ap1+sp2; Y[7]=-ap2+sp4; Y[8]= tp+sp5;
  float tm=sI*tp, am0=sA*ap0, am1=sA*ap1, am2=sA*ap2;
  float sm0=sS*sp0, sm1=sS*sp1, sm2=sS*sp2, sm3=sS*sp3, sm4=sS*sp4, sm5=sS*sp5;
  M[0]= tm+sm0;  M[1]= am0+sm1; M[2]= am1+sm2;
  M[3]=-am0+sm1; M[4]= tm+sm3;  M[5]= am2+sm4;
  M[6]=-am1+sm2; M[7]=-am2+sm4; M[8]= tm+sm5;

  // AB = M@Y + Y@M
  float AB[9];
  #pragma unroll
  for (int i = 0; i < 3; ++i)
    #pragma unroll
    for (int j = 0; j < 3; ++j) {
      float acc = 0.0f;
      #pragma unroll
      for (int k = 0; k < 3; ++k)
        acc += M[i*3+k]*Y[k*3+j] + Y[i*3+k]*M[k*3+j];
      AB[i*3+j] = acc;
    }

  float t2 = (AB[0] + AB[4] + AB[8]) * (1.0f / 3.0f);
  float nab = 0.0f;
  #pragma unroll
  for (int i = 0; i < 9; ++i) nab += AB[i]*AB[i];   // I2+A2+S2 == AB
  float inv2 = 1.0f / (nab + 1.0f);

  __syncthreads();
  {
    float* cp = comps[nl][u];
    cp[0] = t2 * inv2;
    cp[1] = 0.5f*(AB[1]-AB[3]) * inv2;
    cp[2] = 0.5f*(AB[2]-AB[6]) * inv2;
    cp[3] = 0.5f*(AB[5]-AB[7]) * inv2;
    cp[4] = (AB[0]-t2) * inv2;
    cp[5] = 0.5f*(AB[1]+AB[3]) * inv2;
    cp[6] = 0.5f*(AB[2]+AB[6]) * inv2;
    cp[7] = (AB[4]-t2) * inv2;
    cp[8] = 0.5f*(AB[5]+AB[7]) * inv2;
    cp[9] = (AB[8]-t2) * inv2;
  }
  __syncthreads();

  // phase 3: second channel mix (Wt3..5)
  float tq=0.f, aq0=0.f, aq1=0.f, aq2=0.f;
  float sq0=0.f, sq1=0.f, sq2=0.f, sq3=0.f, sq4=0.f, sq5=0.f;
  #pragma unroll 4
  for (int f = 0; f < UNITS; ++f) {
    float w3 = Wt3[u * UNITS + f];
    float w4 = Wt4[u * UNITS + f];
    float w5 = Wt5[u * UNITS + f];
    const float* cf = comps[nl][f];
    tq  += w3 * cf[0];
    aq0 += w4 * cf[1]; aq1 += w4 * cf[2]; aq2 += w4 * cf[3];
    sq0 += w5 * cf[4]; sq1 += w5 * cf[5]; sq2 += w5 * cf[6];
    sq3 += w5 * cf[7]; sq4 += w5 * cf[8]; sq5 += w5 * cf[9];
  }

  float dX[9];
  dX[0]= tq+sq0;  dX[1]= aq0+sq1; dX[2]= aq1+sq2;
  dX[3]=-aq0+sq1; dX[4]= tq+sq3;  dX[5]= aq2+sq4;
  dX[6]=-aq1+sq2; dX[7]=-aq2+sq4; dX[8]= tq+sq5;

  float dd[9];
  #pragma unroll
  for (int i = 0; i < 3; ++i)
    #pragma unroll
    for (int j = 0; j < 3; ++j) {
      float acc = 0.0f;
      #pragma unroll
      for (int k = 0; k < 3; ++k) acc += dX[i*3+k]*dX[k*3+j];
      dd[i*3+j] = acc;
    }

  if (valid) {
    float* op = out + ((size_t)n * UNITS + u) * 9;
    #pragma unroll
    for (int i = 0; i < 9; ++i) op[i] = xns[nl][u][i] + dX[i] + dd[i];
  }
}

// ---------------------------------------------------------------------------
extern "C" void kernel_launch(void* const* d_in, const int* in_sizes, int n_in,
                              void* d_out, int out_size, void* d_ws, size_t ws_size,
                              hipStream_t stream) {
  (void)n_in; (void)out_size; (void)ws_size;
  const float* X         = (const float*)d_in[0];
  const float* bond_dist = (const float*)d_in[1];
  const float* edge_attr = (const float*)d_in[2];
  const int*   edge_dst  = (const int*)  d_in[3];
  const float* Ws1 = (const float*)d_in[4];  const float* bs1 = (const float*)d_in[5];
  const float* Ws2 = (const float*)d_in[6];  const float* bs2 = (const float*)d_in[7];
  const float* Ws3 = (const float*)d_in[8];  const float* bs3 = (const float*)d_in[9];
  const float* Wt0 = (const float*)d_in[10];
  const float* Wt1 = (const float*)d_in[11];
  const float* Wt2 = (const float*)d_in[12];
  const float* Wt3 = (const float*)d_in[13];
  const float* Wt4 = (const float*)d_in[14];
  const float* Wt5 = (const float*)d_in[15];
  float* out = (float*)d_out;

  const int N = in_sizes[0] / (UNITS * 9);
  const int E = in_sizes[1];

  // workspace layout: sums[N][192] f32 | contiguous packed B (f16)
  char* ws = (char*)d_ws;
  float* sums = (float*)ws;
  size_t o = ((size_t)N * H3 * sizeof(float) + 255) & ~(size_t)255;
  _Float16* Ball = (_Float16*)(ws + o);
  _Float16* Bp1 = Ball;
  _Float16* Bp2 = Ball + B1_HALFS;
  _Float16* Bp3 = Ball + B1_HALFS + B2_HALFS;

  zero_f32_kernel<<<(N * H3 + 255) / 256, 256, 0, stream>>>(sums, N * H3);
  pack_b_kernel<<<(B1_HALFS + 255) / 256, 256, 0, stream>>>(Ws1, Bp1, 4, 1, 32);
  pack_b_kernel<<<(B2_HALFS + 255) / 256, 256, 0, stream>>>(Ws2, Bp2, 8, 2, 64);
  pack_b_kernel<<<(B3_HALFS + 255) / 256, 256, 0, stream>>>(Ws3, Bp3, 12, 4, 128);

  size_t smem_bytes = SMEM_HALFS * sizeof(_Float16);   // ~116 KB dynamic LDS
  edge_mlp_kernel<<<(E + EW * SUB * 16 - 1) / (EW * SUB * 16), 128, smem_bytes, stream>>>(
      edge_attr, bond_dist, edge_dst, Ball, bs1, bs2, bs3, sums, E);

  node_kernel<<<(N + 3) / 4, 256, 0, stream>>>(
      X, sums, Wt0, Wt1, Wt2, Wt3, Wt4, Wt5, out, N);
}
